// BlurContrastiveModelPair_13091060318580
// MI455X (gfx1250) — compile-verified, matrix operand order checked by reading, not run on hardware
//
#include <hip/hip_runtime.h>
#include <hip/hip_bf16.h>

// ---------------------------------------------------------------------------
// BlurContrastiveModelPair for MI455X (gfx1250, wave32, WMMA)
//   blur/pack -> f16 transposed weights -> 3x WMMA GEMM (MLP, M=8192) ->
//   normalize -> WMMA Gram->cdist (per batch) ->
//   conv1 (VALU, ch-last f16 out) -> conv2 (WMMA implicit GEMM over 25 taps)
//   -> conv3 (+residual) -> fused softmax/rowsums -> dis
// ---------------------------------------------------------------------------

typedef __attribute__((ext_vector_type(16))) _Float16 v16h;
typedef __attribute__((ext_vector_type(8)))  float    v8f;

static constexpr int kB    = 8;
static constexpr int kT    = 512;
static constexpr int kDIN  = 512;
static constexpr int kDMID = 512;
static constexpr int kDOUT = 128;
static constexpr int kRowsQ = kB * kT;        // 4096
static constexpr int kRowsA = 2 * kRowsQ;     // 8192 (q rows then k rows)
static constexpr float kK0 = 0.15f, kK1 = 0.7f, kK2 = 0.15f;

// ---- workspace layout (bytes) ----
static constexpr size_t OFF_XH   = 0;                                 // 8192x512 f16
static constexpr size_t OFF_HH   = OFF_XH   + (size_t)kRowsA*kDMID*2; // 8192x512 f16
static constexpr size_t OFF_W1T  = OFF_HH   + (size_t)kRowsA*kDMID*2; // [512][512] f16 (N,K)
static constexpr size_t OFF_W2T  = OFF_W1T  + (size_t)kDMID*kDIN*2;
static constexpr size_t OFF_W3T  = OFF_W2T  + (size_t)kDMID*kDMID*2;  // [128][512] f16
static constexpr size_t OFF_E    = OFF_W3T  + (size_t)kDOUT*kDMID*2;  // 8192x128 f32
static constexpr size_t OFF_EH   = OFF_E    + (size_t)kRowsA*kDOUT*4; // 8192x128 f16
static constexpr size_t OFF_SQ   = OFF_EH   + (size_t)kRowsA*kDOUT*2; // 8192 f32
static constexpr size_t OFF_R    = OFF_SQ   + (size_t)kRowsA*4;
static constexpr size_t OFF_AR   = OFF_R    + (size_t)kRowsQ*4;
static constexpr size_t OFF_D    = OFF_AR   + (size_t)kRowsQ*4;       // 8x512x512 f32
static constexpr size_t OFF_ACT1 = OFF_D    + (size_t)kB*kT*kT*4;     // [512][512][32] f16
static constexpr size_t OFF_ACT2 = OFF_ACT1 + (size_t)kT*kT*32*2;     // [512][512][32] f16
static constexpr size_t OFF_WPB  = OFF_ACT2 + (size_t)kT*kT*32*2;     // 25*2*32*16 f16
static constexpr size_t OFF_AF   = OFF_WPB  + (size_t)25*2*32*16*2;   // [512][512] f32
static constexpr size_t OFF_RS   = OFF_AF   + (size_t)kT*kT*4;        // 4096 f32
static constexpr size_t OFF_RDS  = OFF_RS   + (size_t)kRowsQ*4;       // 4096 f32

// ---- output layout (floats) ----
static constexpr size_t OUT_S   = 0;                          // 8*512*512
static constexpr size_t OUT_DIS = OUT_S + (size_t)kB*kT*kT;   // 8
static constexpr size_t OUT_S2B = OUT_DIS + kB;               // 8*512*512
static constexpr size_t OUT_LEN = OUT_S2B + (size_t)kB*kT*kT; // 8

static __device__ __forceinline__ v8f wmma_f16(v16h a, v16h b, v8f c) {
    return __builtin_amdgcn_wmma_f32_16x16x32_f16(false, a, false, b, (short)0, c,
                                                  false, false);
}

// ---------------------------------------------------------------------------
template<int NT>
__device__ __forceinline__ float blockReduceSum(float v, volatile float* red, int tid) {
    red[tid] = v; __syncthreads();
#pragma unroll
    for (int s = NT / 2; s > 0; s >>= 1) {
        if (tid < s) red[tid] = red[tid] + red[tid + s];
        __syncthreads();
    }
    float r = red[0]; __syncthreads();
    return r;
}

template<int NT>
__device__ __forceinline__ float blockReduceMax(float v, volatile float* red, int tid) {
    red[tid] = v; __syncthreads();
#pragma unroll
    for (int s = NT / 2; s > 0; s >>= 1) {
        if (tid < s) red[tid] = fmaxf(red[tid], red[tid + s]);
        __syncthreads();
    }
    float r = red[0]; __syncthreads();
    return r;
}

// ---------------------------------------------------------------------------
// blur + f16 packing of [seq ; avged_seq], plus R / avged_R.  grid (T,B) blk 256
// ---------------------------------------------------------------------------
__global__ void blur_pack_kernel(const float* __restrict__ seq,
                                 const int* __restrict__ len_seq,
                                 _Float16* __restrict__ Xh,
                                 float* __restrict__ R,
                                 float* __restrict__ avgR) {
    const int t = blockIdx.x, b = blockIdx.y, tid = threadIdx.x;
    const int len = len_seq[b];
    const int outlen = len - 2;
    const float* srow = seq + ((size_t)b * kT + t) * kDIN;
    _Float16* xq = Xh + ((size_t)b * kT + t) * kDIN;
    _Float16* xk = Xh + ((size_t)kRowsQ + (size_t)b * kT + t) * kDIN;
    const bool doBlur = (t < kT - 2) && (t < outlen);
    for (int d = tid; d < kDIN; d += 256) {
        const float s0 = srow[d];
        xq[d] = (_Float16)s0;
        float v = 0.0f;
        if (doBlur)
            v = kK0 * s0 + kK1 * srow[kDIN + d] + kK2 * srow[2 * kDIN + d];
        xk[d] = (_Float16)v;
    }
    if (tid == 0) {
        R[b * kT + t]    = (t < len)    ? (t + 1.0f) / (float)len    : 0.0f;
        avgR[b * kT + t] = (t < outlen) ? (t + 1.0f) / (float)outlen : 0.0f;
    }
}

// ---------------------------------------------------------------------------
// SeqtoBlur diagonals + avged_len.  grid (T,B) blk 256
// ---------------------------------------------------------------------------
__global__ void seq2blur_kernel(const int* __restrict__ len_seq,
                                float* __restrict__ out) {
    const int r = blockIdx.x, b = blockIdx.y, tid = threadIdx.x;
    const int outlen = len_seq[b] - 2;
    float* o = out + OUT_S2B + ((size_t)b * kT + r) * kT;
    for (int c = tid; c < kT; c += 256) {
        float v = 0.0f;
        if (c == r)          v = kK0;
        else if (c == r - 1) v = kK1;
        else if (c == r - 2) v = kK2;
        o[c] = (c < outlen) ? v : 0.0f;
    }
    if (r == 0 && tid == 0) out[OUT_LEN + b] = (float)outlen;
}

// ---------------------------------------------------------------------------
// f32 [K][N] -> f16 [N][K] tiled transpose.  grid (N/32, K/32) blk (32,8)
// ---------------------------------------------------------------------------
__global__ void transpose_w_kernel(const float* __restrict__ in,
                                   _Float16* __restrict__ outT,
                                   int K, int N) {
    __shared__ _Float16 tile[32][33];
    const int tx = threadIdx.x, ty = threadIdx.y;
    const int nb = blockIdx.x * 32, kb = blockIdx.y * 32;
#pragma unroll
    for (int j = 0; j < 4; ++j) {
        const int yy = ty + j * 8;
        tile[yy][tx] = (_Float16)in[(size_t)(kb + yy) * N + nb + tx];
    }
    __syncthreads();
#pragma unroll
    for (int j = 0; j < 4; ++j) {
        const int yy = ty + j * 8;
        outT[(size_t)(nb + yy) * K + kb + tx] = tile[tx][yy];
    }
}

// ---------------------------------------------------------------------------
// Pre-pack conv2 weights into WMMA B-fragment register order:
//   WpB[((tap*2+nt)*32 + lane)*16 + j] with lane=(hi<<4)|lr, co=nt*16+lr,
//   ci -> (hi=(ci>>3)&1, j=(ci&7)+((ci>>4)<<3)) per CDNA5 16-bit B layout.
// ---------------------------------------------------------------------------
__global__ void prepack_wpb_kernel(const float* __restrict__ w, // [30][30][5][5]
                                   _Float16* __restrict__ WpB) {
    const int idx = blockIdx.x * 256 + threadIdx.x;   // 25*32*32 = 25600
    if (idx >= 25 * 32 * 32) return;
    const int tap = idx >> 10;
    const int co  = (idx >> 5) & 31;
    const int ci  = idx & 31;
    float v = 0.0f;
    if (co < 30 && ci < 30) v = w[(co * 30 + ci) * 25 + tap];
    const int nt   = co >> 4;
    const int lr   = co & 15;
    const int hi   = (ci >> 3) & 1;
    const int j    = (ci & 7) + ((ci >> 4) << 3);
    const int lane = (hi << 4) | lr;
    WpB[(((size_t)tap * 2 + nt) * 32 + lane) * 16 + j] = (_Float16)v;
}

// ---------------------------------------------------------------------------
// WMMA GEMM: C[M x N] = epilogue(A[M x K] @ B^T + bias), B given [N][K] f16.
//   MODE 0: f16 out, bias + relu   MODE 1: f32 out, bias   MODE 2: cdist -> D
// Block 256 = 8 waves; tile 128x128x32; wave does 32x64 (2x4 WMMA tiles).
// ---------------------------------------------------------------------------
template<int MODE>
__global__ __launch_bounds__(256)
void wmma_gemm_kernel(const _Float16* __restrict__ A,
                      const _Float16* __restrict__ Bm,
                      int N, int K,
                      long long aBatch, long long bBatch, long long oBatch,
                      const float* __restrict__ bias,
                      _Float16* __restrict__ outH,
                      float* __restrict__ outF,
                      const float* __restrict__ sqQ,
                      const float* __restrict__ sqK) {
    __shared__ __align__(16) _Float16 As[128][40];  // [m][k] +8 pad
    __shared__ __align__(16) _Float16 Bs[128][40];  // [n][k] +8 pad

    const int tid  = threadIdx.x;
    const int lane = tid & 31;
    const int wave = tid >> 5;
    const int wm   = wave >> 1;
    const int wn   = wave & 1;
    const int lr   = lane & 15;
    const int hi   = lane >> 4;

    const int rowBase = blockIdx.y * 128;
    const int colBase = blockIdx.x * 128;
    const _Float16* Ab = A  + (size_t)blockIdx.z * (size_t)aBatch;
    const _Float16* Bb = Bm + (size_t)blockIdx.z * (size_t)bBatch;

    v8f acc[2][4];
#pragma unroll
    for (int mt = 0; mt < 2; ++mt)
#pragma unroll
        for (int nt = 0; nt < 4; ++nt)
            acc[mt][nt] = v8f{};

    const int r0 = tid >> 2;
    const int c8 = (tid & 3) * 8;

    for (int kt = 0; kt < K; kt += 32) {
        __syncthreads();
#pragma unroll
        for (int rr = 0; rr < 2; ++rr) {
            const int r = r0 + rr * 64;
            *(uint4*)(&As[r][c8]) =
                *(const uint4*)(Ab + (size_t)(rowBase + r) * K + kt + c8);
            *(uint4*)(&Bs[r][c8]) =
                *(const uint4*)(Bb + (size_t)(colBase + r) * K + kt + c8);
        }
        if (kt + 32 < K) {
            __builtin_prefetch(Ab + (size_t)(rowBase + r0) * K + kt + 32, 0, 1);
            __builtin_prefetch(Bb + (size_t)(colBase + r0) * K + kt + 32, 0, 1);
        }
        __syncthreads();

        v16h aF[2];
#pragma unroll
        for (int mt = 0; mt < 2; ++mt) {
            const _Float16* p = &As[wm * 32 + mt * 16 + lr][hi * 8];
            ((uint4*)&aF[mt])[0] = *(const uint4*)(p);
            ((uint4*)&aF[mt])[1] = *(const uint4*)(p + 16);
        }
#pragma unroll
        for (int nt = 0; nt < 4; ++nt) {
            v16h bF;
            const _Float16* q = &Bs[wn * 64 + nt * 16 + lr][hi * 8];
            ((uint4*)&bF)[0] = *(const uint4*)(q);
            ((uint4*)&bF)[1] = *(const uint4*)(q + 16);
#pragma unroll
            for (int mt = 0; mt < 2; ++mt)
                acc[mt][nt] = wmma_f16(aF[mt], bF, acc[mt][nt]);
        }
    }

    const int colIn = lane & 15;
    const int rOff  = 8 * (lane >> 4);
#pragma unroll
    for (int mt = 0; mt < 2; ++mt) {
#pragma unroll
        for (int nt = 0; nt < 4; ++nt) {
            const int gr0 = rowBase + wm * 32 + mt * 16 + rOff;
            const int gc  = colBase + wn * 64 + nt * 16 + colIn;
            const float bv = (MODE == 2) ? 0.0f : bias[gc];
#pragma unroll
            for (int i = 0; i < 8; ++i) {
                const int row = gr0 + i;
                float v = acc[mt][nt][i];
                if (MODE == 0) {
                    v += bv; v = v > 0.0f ? v : 0.0f;
                    outH[(size_t)blockIdx.z * (size_t)oBatch + (size_t)row * N + gc] = (_Float16)v;
                } else if (MODE == 1) {
                    v += bv;
                    outF[(size_t)blockIdx.z * (size_t)oBatch + (size_t)row * N + gc] = v;
                } else {
                    float s = sqQ[blockIdx.z * kT + row] + sqK[blockIdx.z * kT + gc] - 2.0f * v;
                    s = s < 1e-12f ? 1e-12f : s;
                    outF[(size_t)blockIdx.z * (size_t)oBatch + (size_t)row * N + gc] = sqrtf(s);
                }
            }
        }
    }
}

// ---------------------------------------------------------------------------
// L2-normalize 128-dim rows, nan/inf -> 0; emit f16 copy + sum-of-squares
// ---------------------------------------------------------------------------
__global__ void normalize_kernel(const float* __restrict__ E,
                                 _Float16* __restrict__ Eh,
                                 float* __restrict__ sq) {
    __shared__ float red[128];
    const int row = blockIdx.x, tid = threadIdx.x;
    const float v = E[(size_t)row * kDOUT + tid];
    const float ss = blockReduceSum<128>(v * v, red, tid);
    float o = v / sqrtf(ss);
    if (!__builtin_isfinite(o)) o = 0.0f;
    const float s2 = blockReduceSum<128>(o * o, red, tid);
    if (tid == 0) sq[row] = s2;
    Eh[(size_t)row * kDOUT + tid] = (_Float16)o;
}

// ---------------------------------------------------------------------------
// conv1: {D, |R-avgR|} (2ch) -> 30ch 5x5 pad2 relu, channels-last f16 out
// grid (32,32) blk (16,16)
// ---------------------------------------------------------------------------
__global__ __launch_bounds__(256)
void conv1_kernel(const float* __restrict__ Dp,
                  const float* __restrict__ Rq,
                  const float* __restrict__ Rk,
                  const float* __restrict__ w,    // [30][2][5][5]
                  const float* __restrict__ bias, // [30]
                  _Float16* __restrict__ out) {   // [512][512][32]
    __shared__ float t0[20][20], t1[20][20];
    __shared__ float wl[1500];
    const int tx = threadIdx.x, ty = threadIdx.y;
    const int tid = ty * 16 + tx;
    for (int i = tid; i < 1500; i += 256) wl[i] = w[i];
    const int gx0 = blockIdx.x * 16 - 2;
    const int gy0 = blockIdx.y * 16 - 2;
    for (int i = tid; i < 400; i += 256) {
        const int yy = i / 20, xx = i % 20;
        const int gy = gy0 + yy, gx = gx0 + xx;
        const bool in = (gy >= 0 && gy < kT && gx >= 0 && gx < kT);
        t0[yy][xx] = in ? Dp[(size_t)gy * kT + gx] : 0.0f;
        t1[yy][xx] = in ? fabsf(Rq[gy] - Rk[gx]) : 0.0f;
    }
    __syncthreads();
    float acc[30];
#pragma unroll
    for (int c = 0; c < 30; ++c) acc[c] = 0.0f;
    for (int ci = 0; ci < 2; ++ci) {
        for (int dy = 0; dy < 5; ++dy)
            for (int dx = 0; dx < 5; ++dx) {
                const float v = (ci == 0) ? t0[ty + dy][tx + dx] : t1[ty + dy][tx + dx];
                const int tap = ci * 25 + dy * 5 + dx;
#pragma unroll
                for (int co = 0; co < 30; ++co)
                    acc[co] += v * wl[co * 50 + tap];
            }
    }
    const int y = blockIdx.y * 16 + ty, x = blockIdx.x * 16 + tx;
    __align__(16) _Float16 ob[32];
#pragma unroll
    for (int co = 0; co < 30; ++co) {
        const float v = acc[co] + bias[co];
        ob[co] = (_Float16)(v > 0.0f ? v : 0.0f);
    }
    ob[30] = (_Float16)0.0f; ob[31] = (_Float16)0.0f;
    _Float16* op = out + ((size_t)y * kT + x) * 32;
#pragma unroll
    for (int j = 0; j < 4; ++j)
        ((uint4*)op)[j] = ((const uint4*)ob)[j];
}

// ---------------------------------------------------------------------------
// conv2 as WMMA implicit GEMM: out[p,co] = relu(b + sum_tap A_tap[p,ci]*W[tap][ci,co])
// Activations channels-last f16 [512][512][32]; K=32 per tap; 25 taps.
// Block 256 = 8 waves; output tile 16 rows x 32 cols; wave: 4 M-tiles x 2 N-tiles.
// grid (512/32, 512/16)
// ---------------------------------------------------------------------------
__global__ __launch_bounds__(256)
void conv2_wmma_kernel(const _Float16* __restrict__ act,  // [512][512][32]
                       const _Float16* __restrict__ WpB,  // B fragments
                       const float* __restrict__ bias,    // [30]
                       _Float16* __restrict__ out) {      // [512][512][32]
    // pixel slots padded 32 -> 40 halves (stride 20 banks: conflict-free)
    __shared__ __align__(16) _Float16 tin[20][36][40];    // 57.6 KB
    const int tid  = threadIdx.x;
    const int lane = tid & 31;
    const int wave = tid >> 5;
    const int lr   = lane & 15;
    const int hi   = lane >> 4;
    const int x0 = blockIdx.x * 32;
    const int y0 = blockIdx.y * 16;

    for (int p = tid; p < 20 * 36; p += 256) {
        const int yy = p / 36, xx = p % 36;
        const int gy = y0 + yy - 2, gx = x0 + xx - 2;
        uint4 v0{}, v1{}, v2{}, v3{};
        if (gy >= 0 && gy < kT && gx >= 0 && gx < kT) {
            const uint4* src = (const uint4*)(act + ((size_t)gy * kT + gx) * 32);
            v0 = src[0]; v1 = src[1]; v2 = src[2]; v3 = src[3];
        }
        uint4* dst = (uint4*)(&tin[yy][xx][0]);
        dst[0] = v0; dst[1] = v1; dst[2] = v2; dst[3] = v3;
    }
    __syncthreads();

    v8f acc[4][2];
#pragma unroll
    for (int mi = 0; mi < 4; ++mi) {
        acc[mi][0] = v8f{};
        acc[mi][1] = v8f{};
    }

    for (int tap = 0; tap < 25; ++tap) {
        const int dy = tap / 5, dx = tap % 5;
        v16h bF[2];
#pragma unroll
        for (int nt = 0; nt < 2; ++nt) {
            const _Float16* wp = WpB + (((size_t)tap * 2 + nt) * 32 + lane) * 16;
            ((uint4*)&bF[nt])[0] = *(const uint4*)(wp);
            ((uint4*)&bF[nt])[1] = *(const uint4*)(wp + 8);
        }
#pragma unroll
        for (int mi = 0; mi < 4; ++mi) {
            const int py = wave * 2 + (mi >> 1);
            const int g  = mi & 1;
            const _Float16* ap = &tin[py + dy][g * 16 + lr + dx][hi * 8];
            v16h aF;
            ((uint4*)&aF)[0] = *(const uint4*)(ap);
            ((uint4*)&aF)[1] = *(const uint4*)(ap + 16);
            acc[mi][0] = wmma_f16(aF, bF[0], acc[mi][0]);
            acc[mi][1] = wmma_f16(aF, bF[1], acc[mi][1]);
        }
    }

#pragma unroll
    for (int mi = 0; mi < 4; ++mi) {
        const int py = wave * 2 + (mi >> 1);
        const int g  = mi & 1;
#pragma unroll
        for (int nt = 0; nt < 2; ++nt) {
            const int co = nt * 16 + (lane & 15);
            const float bv = (co < 30) ? bias[co] : 0.0f;
#pragma unroll
            for (int i = 0; i < 8; ++i) {
                const int m = i + 8 * (lane >> 4);
                const int x = x0 + g * 16 + m;
                const int y = y0 + py;
                float v = acc[mi][nt][i] + bv;
                v = v > 0.0f ? v : 0.0f;
                out[((size_t)y * kT + x) * 32 + co] =
                    (co < 30) ? (_Float16)v : (_Float16)0.0f;
            }
        }
    }
}

// ---------------------------------------------------------------------------
// conv3: 30ch (channels-last f16) -> 1ch, 3x3 pad1, + c3b + D residual
// grid (32,32) blk (16,16)
// ---------------------------------------------------------------------------
__global__ __launch_bounds__(256)
void conv3_kernel(const _Float16* __restrict__ act,  // [512][512][32]
                  const float* __restrict__ w,       // [1][30][3][3]
                  const float* __restrict__ bias,
                  const float* __restrict__ Dp,      // [512][512]
                  float* __restrict__ Af) {          // [512][512]
    __shared__ float wl[270];
    const int tx = threadIdx.x, ty = threadIdx.y;
    const int tid = ty * 16 + tx;
    for (int i = tid; i < 270; i += 256) wl[i] = w[i];
    __syncthreads();
    const int y = blockIdx.y * 16 + ty, x = blockIdx.x * 16 + tx;
    float acc = 0.0f;
#pragma unroll
    for (int dy = 0; dy < 3; ++dy) {
#pragma unroll
        for (int dx = 0; dx < 3; ++dx) {
            const int gy = y + dy - 1, gx = x + dx - 1;
            if (gy >= 0 && gy < kT && gx >= 0 && gx < kT) {
                const _Float16* p = act + ((size_t)gy * kT + gx) * 32;
#pragma unroll
                for (int ci = 0; ci < 30; ++ci)
                    acc += (float)p[ci] * wl[ci * 9 + dy * 3 + dx];
            }
        }
    }
    Af[(size_t)y * kT + x] = acc + bias[0] + Dp[(size_t)y * kT + x];
}

// ---------------------------------------------------------------------------
// softmax over k per row; write S; emit per-row sums of S and D*S
// ---------------------------------------------------------------------------
__global__ void softmax_kernel(const float* __restrict__ Af,
                               const float* __restrict__ Dp,
                               float* __restrict__ Sout,
                               float* __restrict__ rowS,
                               float* __restrict__ rowDS) {
    __shared__ float red[256];
    const int q = blockIdx.x, tid = threadIdx.x;
    const float x0 = -Af[(size_t)q * kT + tid];
    const float x1 = -Af[(size_t)q * kT + tid + 256];
    const float M  = blockReduceMax<256>(fmaxf(x0, x1), red, tid);
    const float e0 = expf(x0 - M), e1 = expf(x1 - M);
    const float sum = blockReduceSum<256>(e0 + e1, red, tid);
    const float inv = 1.0f / sum;
    const float s0 = e0 * inv, s1 = e1 * inv;
    Sout[(size_t)q * kT + tid]       = s0;
    Sout[(size_t)q * kT + tid + 256] = s1;
    const float rs  = blockReduceSum<256>(s0 + s1, red, tid);
    const float rds = blockReduceSum<256>(Dp[(size_t)q * kT + tid] * s0 +
                                          Dp[(size_t)q * kT + tid + 256] * s1, red, tid);
    if (tid == 0) { rowS[q] = rs; rowDS[q] = rds; }
}

// ---------------------------------------------------------------------------
__global__ void dis_kernel(const float* __restrict__ rowS,
                           const float* __restrict__ rowDS,
                           float* __restrict__ disOut) {
    __shared__ float red[256];
    const int b = blockIdx.x, tid = threadIdx.x;
    float s = 0.0f, ds = 0.0f;
    for (int i = tid; i < kT; i += 256) {
        s  += rowS[b * kT + i];
        ds += rowDS[b * kT + i];
    }
    const float ts  = blockReduceSum<256>(s, red, tid);
    const float tds = blockReduceSum<256>(ds, red, tid);
    if (tid == 0) disOut[b] = tds / ts;
}

// ---------------------------------------------------------------------------
extern "C" void kernel_launch(void* const* d_in, const int* in_sizes, int n_in,
                              void* d_out, int out_size, void* d_ws, size_t ws_size,
                              hipStream_t stream) {
    const float* seq     = (const float*)d_in[0];
    const int*   len_seq = (const int*)d_in[1];
    const float* W1 = (const float*)d_in[2];  const float* b1 = (const float*)d_in[3];
    const float* W2 = (const float*)d_in[4];  const float* b2 = (const float*)d_in[5];
    const float* W3 = (const float*)d_in[6];  const float* b3 = (const float*)d_in[7];
    const float* c1w = (const float*)d_in[8];  const float* c1b = (const float*)d_in[9];
    const float* c2w = (const float*)d_in[10]; const float* c2b = (const float*)d_in[11];
    const float* c3w = (const float*)d_in[12]; const float* c3b = (const float*)d_in[13];
    float* out = (float*)d_out;

    char* ws = (char*)d_ws;
    _Float16* Xh  = (_Float16*)(ws + OFF_XH);
    _Float16* Hh  = (_Float16*)(ws + OFF_HH);
    _Float16* w1t = (_Float16*)(ws + OFF_W1T);
    _Float16* w2t = (_Float16*)(ws + OFF_W2T);
    _Float16* w3t = (_Float16*)(ws + OFF_W3T);
    float*    E   = (float*)(ws + OFF_E);
    _Float16* Eh  = (_Float16*)(ws + OFF_EH);
    float*    sq  = (float*)(ws + OFF_SQ);
    float*    R   = (float*)(ws + OFF_R);
    float*    aR  = (float*)(ws + OFF_AR);
    float*    Dbf = (float*)(ws + OFF_D);
    _Float16* a1  = (_Float16*)(ws + OFF_ACT1);
    _Float16* a2  = (_Float16*)(ws + OFF_ACT2);
    _Float16* WpB = (_Float16*)(ws + OFF_WPB);
    float*    Af  = (float*)(ws + OFF_AF);
    float*    rS  = (float*)(ws + OFF_RS);
    float*    rDS = (float*)(ws + OFF_RDS);

    // stage 0: packing / masks / weight repacks
    blur_pack_kernel<<<dim3(kT, kB), 256, 0, stream>>>(seq, len_seq, Xh, R, aR);
    seq2blur_kernel<<<dim3(kT, kB), 256, 0, stream>>>(len_seq, out);
    transpose_w_kernel<<<dim3(kDMID / 32, kDIN / 32), dim3(32, 8), 0, stream>>>(W1, w1t, kDIN, kDMID);
    transpose_w_kernel<<<dim3(kDMID / 32, kDMID / 32), dim3(32, 8), 0, stream>>>(W2, w2t, kDMID, kDMID);
    transpose_w_kernel<<<dim3(kDOUT / 32, kDMID / 32), dim3(32, 8), 0, stream>>>(W3, w3t, kDMID, kDOUT);
    prepack_wpb_kernel<<<(25 * 32 * 32 + 255) / 256, 256, 0, stream>>>(c2w, WpB);

    // stage 1: MLP (M=8192 combined q|k), WMMA f16
    wmma_gemm_kernel<0><<<dim3(kDMID / 128, kRowsA / 128, 1), 256, 0, stream>>>(
        Xh, w1t, kDMID, kDIN, 0, 0, 0, b1, Hh, nullptr, nullptr, nullptr);
    wmma_gemm_kernel<0><<<dim3(kDMID / 128, kRowsA / 128, 1), 256, 0, stream>>>(
        Hh, w2t, kDMID, kDMID, 0, 0, 0, b2, Xh, nullptr, nullptr, nullptr);
    wmma_gemm_kernel<1><<<dim3(kDOUT / 128, kRowsA / 128, 1), 256, 0, stream>>>(
        Xh, w3t, kDOUT, kDMID, 0, 0, 0, b3, nullptr, E, nullptr, nullptr);
    normalize_kernel<<<kRowsA, 128, 0, stream>>>(E, Eh, sq);

    // stage 2: Gram -> cdist D (per batch), WMMA f16
    wmma_gemm_kernel<2><<<dim3(kT / 128, kT / 128, kB), 256, 0, stream>>>(
        Eh, Eh + (size_t)kRowsQ * kDOUT, kT, kDOUT,
        (long long)kT * kDOUT, (long long)kT * kDOUT, (long long)kT * kT,
        nullptr, nullptr, Dbf, sq, sq + kRowsQ);

    // stage 3: conv stack + softmax, per batch
    for (int b = 0; b < kB; ++b) {
        const float* Dp = Dbf + (size_t)b * kT * kT;
        conv1_kernel<<<dim3(32, 32), dim3(16, 16), 0, stream>>>(
            Dp, R + b * kT, aR + b * kT, c1w, c1b, a1);
        conv2_wmma_kernel<<<dim3(kT / 32, kT / 16), 256, 0, stream>>>(a1, WpB, c2b, a2);
        conv3_kernel<<<dim3(32, 32), dim3(16, 16), 0, stream>>>(a2, c3w, c3b, Dp, Af);
        softmax_kernel<<<kT, 256, 0, stream>>>(
            Af, Dp, out + OUT_S + (size_t)b * kT * kT, rS + b * kT, rDS + b * kT);
    }
    dis_kernel<<<kB, 256, 0, stream>>>(rS, rDS, out + OUT_DIS);
}
